// QuantumCollapseInference_75857712382445
// MI455X (gfx1250) — matrix operand order repaired and checked by reading, not compiled, not run on hardware
//
#include <hip/hip_runtime.h>

typedef __bf16 bf16;
typedef __attribute__((ext_vector_type(16))) __bf16 v16bf;
typedef __attribute__((ext_vector_type(8)))  __bf16 bf16x8;
typedef __attribute__((ext_vector_type(8)))  float  v8f;
typedef __attribute__((ext_vector_type(4)))  float  f32x4;

static __device__ __forceinline__ v8f wmma_bf16(v16bf a, v16bf b, v8f c) {
  // D = A(16x32 bf16) * B(32x16 bf16) + C(16x16 f32)
  return __builtin_amdgcn_wmma_f32_16x16x32_bf16(false, a, false, b, (short)0, c, false, false);
}

static __device__ __forceinline__ v16bf pack16(bf16x8 lo, bf16x8 hi) {
  v16bf r;
#pragma unroll
  for (int i = 0; i < 8; ++i) { r[i] = lo[i]; r[8 + i] = hi[i]; }
  return r;
}

// Phase-2 A operand (basisT): lane holds 16 *consecutive* K values -> one 32B contiguous region
static __device__ __forceinline__ v16bf loadB32(const bf16* p) {
  return pack16(*(const bf16x8*)p, *(const bf16x8*)(p + 8));
}

// A operand: lane holds two 8-element runs at +0 and +16 (16-bit A layout)
static __device__ __forceinline__ v16bf loadA32(const bf16* p) {
  return pack16(*(const bf16x8*)p, *(const bf16x8*)(p + 16));
}

static __device__ __forceinline__ v8f zero8() {
  v8f z;
#pragma unroll
  for (int i = 0; i < 8; ++i) z[i] = 0.0f;
  return z;
}

// After shfl_xor{1,2,4,8} reduction, lane holds 8 row values (row = v + 8*(lane/16)).
// Returns the value for row (lane & 15).
static __device__ __forceinline__ float gather_row(const float p[8], int lane) {
  float x01 = (lane & 1) ? p[1] : p[0];
  float x23 = (lane & 1) ? p[3] : p[2];
  float x45 = (lane & 1) ? p[5] : p[4];
  float x67 = (lane & 1) ? p[7] : p[6];
  float x03 = (lane & 2) ? x23 : x01;
  float x47 = (lane & 2) ? x67 : x45;
  float xsel = (lane & 4) ? x47 : x03;   // = p[lane & 7]
  int r = lane & 15;
  int src = (r & 8) ? (16 + (r & 7)) : (r & 7);
  return __shfl(xsel, src, 32);
}

// ---------------- prologue: basis -> bf16 (+ transposed + negated-imag), softplus(w)
__global__ void qc_prep(const float* __restrict__ br_f, const float* __restrict__ bi_f,
                        const float* __restrict__ pw,
                        bf16* __restrict__ Wbr,  bf16* __restrict__ Wbi,  bf16* __restrict__ Wbin,
                        bf16* __restrict__ WbrT, bf16* __restrict__ WbiT, bf16* __restrict__ WbinT,
                        float* __restrict__ Wf) {
  int m = blockIdx.x, d = threadIdx.x;
  float vr = br_f[m * 256 + d];
  float vi = bi_f[m * 256 + d];
  Wbr [m * 256 + d] = (bf16)vr;
  Wbi [m * 256 + d] = (bf16)vi;
  Wbin[m * 256 + d] = (bf16)(-vi);
  WbrT [d * 256 + m] = (bf16)vr;
  WbiT [d * 256 + m] = (bf16)vi;
  WbinT[d * 256 + m] = (bf16)(-vi);
  if (m == 0) {
    float x = pw[d];
    Wf[d] = (x > 20.f) ? x : log1pf(expf(x));  // softplus
  }
}

// ---------------- main fused kernel: 4 waves/block, 16 rows/wave
__global__ __launch_bounds__(128)
__attribute__((amdgpu_waves_per_eu(4)))
void qc_main(
    const float* __restrict__ xr, const float* __restrict__ xi,
    const bf16* __restrict__ Wbr,  const bf16* __restrict__ Wbi,  const bf16* __restrict__ Wbin,
    const bf16* __restrict__ WbrT, const bf16* __restrict__ WbiT, const bf16* __restrict__ WbinT,
    const float* __restrict__ Wf,
    float* __restrict__ out, long plane) {
  // per-wave 16x256 complex scratch (bf16), XOR-swizzled columns, 64KB total
  __shared__ bf16 tile[4][2][16][256];

  const int lane = threadIdx.x & 31;
  const int wave = threadIdx.x >> 5;
  const int g = lane >> 4;   // half-wave group
  const int r = lane & 15;   // row within tile (phase1/2) / B column
  const int Xr = (r & 7) << 3;  // column swizzle for row r
  const long R0 = (long)blockIdx.x * 64 + (long)wave * 16;

  // ---- phase 0: load psi as bf16 A-fragments + entropy accumulation (f32) ----
  v16bf aR[8], aI[8];
  float t1 = 0.f, t2 = 0.f;
  {
    const float* prr = xr + (R0 + r) * 256;
    const float* pri = xi + (R0 + r) * 256;
#pragma unroll
    for (int kc = 0; kc < 8; ++kc) {
      int s1 = kc * 32 + g * 8;  // A layout: elems 0..7 -> K=s1.. ; 8..15 -> K=s1+16..
      f32x4 a0 = *(const f32x4*)(prr + s1);
      f32x4 a1 = *(const f32x4*)(prr + s1 + 4);
      f32x4 a2 = *(const f32x4*)(prr + s1 + 16);
      f32x4 a3 = *(const f32x4*)(prr + s1 + 20);
      f32x4 b0 = *(const f32x4*)(pri + s1);
      f32x4 b1 = *(const f32x4*)(pri + s1 + 4);
      f32x4 b2 = *(const f32x4*)(pri + s1 + 16);
      f32x4 b3 = *(const f32x4*)(pri + s1 + 20);
      float fre[16], fim[16];
#pragma unroll
      for (int j = 0; j < 4; ++j) {
        fre[j] = a0[j]; fre[4 + j] = a1[j]; fre[8 + j] = a2[j]; fre[12 + j] = a3[j];
        fim[j] = b0[j]; fim[4 + j] = b1[j]; fim[8 + j] = b2[j]; fim[12 + j] = b3[j];
      }
#pragma unroll
      for (int j = 0; j < 16; ++j) {
        aR[kc][j] = (bf16)fre[j];
        aI[kc][j] = (bf16)fim[j];
        float p = fre[j] * fre[j] + fim[j] * fim[j];
        t1 += p;
        t2 += p * logf(fmaxf(p, 1e-30f));
      }
    }
  }
  t1 += __shfl_xor(t1, 16, 32);
  t2 += __shfl_xor(t2, 16, 32);
  float Pn = t1 + 1e-12f;
  float H = logf(Pn) - t2 / Pn;                           // entropy of normalized p
  float conf = 1.0f / (1.0f + expf((H - 0.5f) * 5.0f));   // sigmoid(-(H-tau)*T), row r

  // ---- phase 1: inner = psi . conj(basis)^T ----
  float rawsum[8];
#pragma unroll
  for (int v = 0; v < 8; ++v) rawsum[v] = 0.f;

#pragma unroll 1
  for (int mt = 0; mt < 16; ++mt) {
    v8f cr = zero8(), ci = zero8();
    const int base = (mt * 16 + r) * 256 + g * 8;  // B: column m=mt*16+r, K=g*8..+7 and g*8+16..+23
#pragma unroll
    for (int kc = 0; kc < 8; ++kc) {
      v16bf bR = loadA32(Wbr + base + kc * 32);
      v16bf bI = loadA32(Wbi + base + kc * 32);
      v16bf bN = loadA32(Wbin + base + kc * 32);
      cr = wmma_bf16(aR[kc], bR, cr);   // + xr*br
      cr = wmma_bf16(aI[kc], bI, cr);   // + xi*bi
      ci = wmma_bf16(aI[kc], bR, ci);   // + xi*br
      ci = wmma_bf16(aR[kc], bN, ci);   // - xr*bi
    }
    float wm = Wf[mt * 16 + r];
#pragma unroll
    for (int v = 0; v < 8; ++v) {
      float re = cr[v], im = ci[v];
      rawsum[v] = fmaf(wm, re * re + im * im, rawsum[v]);
      int col = (mt * 16 + r) ^ (v << 3);   // swizzle with row (v+8g): X = (v&7)<<3
      tile[wave][0][v + 8 * g][col] = (bf16)re;
      tile[wave][1][v + 8 * g][col] = (bf16)im;
    }
  }
#pragma unroll
  for (int v = 0; v < 8; ++v) {
    rawsum[v] += __shfl_xor(rawsum[v], 1, 32);
    rawsum[v] += __shfl_xor(rawsum[v], 2, 32);
    rawsum[v] += __shfl_xor(rawsum[v], 4, 32);
    rawsum[v] += __shfl_xor(rawsum[v], 8, 32);
  }
  float rS = 1.0f / (gather_row(rawsum, lane) + 1e-8f);  // 1/(S+eps) for row r

  // ---- build amp directly as B-fragments (lane = psi-row r, K = m) ----
  v16bf ampR[8], ampI[8];
#pragma unroll
  for (int mc = 0; mc < 8; ++mc) {
    int cb = mc * 32 + g * 16;  // 16 consecutive m values
    const bf16* baseRe = &tile[wave][0][r][0];
    const bf16* baseIm = &tile[wave][1][r][0];
    bf16x8 re1 = *(const bf16x8*)(baseRe + (cb ^ Xr));
    bf16x8 re2 = *(const bf16x8*)(baseRe + ((cb + 8) ^ Xr));
    bf16x8 im1 = *(const bf16x8*)(baseIm + (cb ^ Xr));
    bf16x8 im2 = *(const bf16x8*)(baseIm + ((cb + 8) ^ Xr));
    f32x4 w0 = *(const f32x4*)(Wf + cb);
    f32x4 w1 = *(const f32x4*)(Wf + cb + 4);
    f32x4 w2 = *(const f32x4*)(Wf + cb + 8);
    f32x4 w3 = *(const f32x4*)(Wf + cb + 12);
    float wv[16];
#pragma unroll
    for (int j = 0; j < 4; ++j) {
      wv[j] = w0[j]; wv[4 + j] = w1[j]; wv[8 + j] = w2[j]; wv[12 + j] = w3[j];
    }
#pragma unroll
    for (int j = 0; j < 16; ++j) {
      float re = (float)((j < 8) ? re1[j] : re2[j - 8]);
      float im = (float)((j < 8) ? im1[j] : im2[j - 8]);
      float wj = wv[j];
      float t = wj * (re * re + im * im) * rS;   // probs
      t = fmaxf(t, 1e-8f);
      float sc = sqrtf(wj * t);                  // sqrt(w)*sqrt(probs)
      ampR[mc][j] = (bf16)(sc * re);
      ampI[mc][j] = (bf16)(sc * im);
    }
  }

  // ---- phase 2 (transposed): collapsed^T[d][row] = basisT(A) x amp^T(B) ----
  float nrmv = 0.f;
#pragma unroll 1
  for (int dt = 0; dt < 16; ++dt) {
    v8f cr = zero8(), ci = zero8();
    const int base = (dt * 16 + r) * 256;  // A: row d = dt*16+r, K = m
#pragma unroll
    for (int mc = 0; mc < 8; ++mc) {
      int s1 = mc * 32 + g * 16;
      v16bf Abr = loadB32(WbrT + base + s1);
      v16bf Abi = loadB32(WbiT + base + s1);
      v16bf Abn = loadB32(WbinT + base + s1);
      cr = wmma_bf16(Abr, ampR[mc], cr);   // + br*ar
      cr = wmma_bf16(Abn, ampI[mc], cr);   // - bi*ai
      ci = wmma_bf16(Abi, ampR[mc], ci);   // + bi*ar
      ci = wmma_bf16(Abr, ampI[mc], ci);   // + br*ai
    }
#pragma unroll
    for (int v = 0; v < 8; ++v) {
      float re = cr[v], im = ci[v];          // element (d = dt*16+v+8g, row = r)
      nrmv = fmaf(re, re, fmaf(im, im, nrmv));
      int col = (dt * 16 + v + 8 * g) ^ Xr;  // store collapsed[row r][d], swizzled
      tile[wave][0][r][col] = (bf16)re;
      tile[wave][1][r][col] = (bf16)im;
    }
  }
  nrmv += __shfl_xor(nrmv, 16, 32);          // other half of d-range
  float invn = rsqrtf(fmaxf(nrmv, 1e-12f));  // row r

  // ---- epilogue: blend & store (coalesced float4) ----
#pragma unroll 1
  for (int rr = 0; rr < 16; ++rr) {
    float c  = __shfl(conf, rr, 32);
    float iv = __shfl(invn, rr, 32);
    float omc = 1.0f - c;
    long Rw = R0 + rr;
    const float* pr = xr + Rw * 256 + lane * 8;
    const float* pi = xi + Rw * 256 + lane * 8;
    f32x4 xr0 = *(const f32x4*)(pr);
    f32x4 xr1 = *(const f32x4*)(pr + 4);
    f32x4 xi0 = *(const f32x4*)(pi);
    f32x4 xi1 = *(const f32x4*)(pi + 4);
    int X = (rr & 7) << 3;
    bf16x8 cre = *(const bf16x8*)&tile[wave][0][rr][(lane * 8) ^ X];
    bf16x8 cim = *(const bf16x8*)&tile[wave][1][rr][(lane * 8) ^ X];
    f32x4 oR0, oR1, oI0, oI1;
#pragma unroll
    for (int j = 0; j < 4; ++j) {
      oR0[j] = c * xr0[j] + omc * ((float)cre[j]) * iv;
      oR1[j] = c * xr1[j] + omc * ((float)cre[4 + j]) * iv;
      oI0[j] = c * xi0[j] + omc * ((float)cim[j]) * iv;
      oI1[j] = c * xi1[j] + omc * ((float)cim[4 + j]) * iv;
    }
    float* dre = out + Rw * 256 + lane * 8;
    float* dim = out + plane + Rw * 256 + lane * 8;
    *(f32x4*)(dre)     = oR0;
    *(f32x4*)(dre + 4) = oR1;
    *(f32x4*)(dim)     = oI0;
    *(f32x4*)(dim + 4) = oI1;
  }
}

extern "C" void kernel_launch(void* const* d_in, const int* in_sizes, int n_in,
                              void* d_out, int out_size, void* d_ws, size_t ws_size,
                              hipStream_t stream) {
  const float* x_real     = (const float*)d_in[0];
  const float* x_imag     = (const float*)d_in[1];
  const float* basis_real = (const float*)d_in[2];
  const float* basis_imag = (const float*)d_in[3];
  const float* povm       = (const float*)d_in[4];

  const long rows  = (long)in_sizes[0] / 256;   // B*N = 65536
  const long plane = rows * 256;                // elements per real/imag plane

  // workspace: 6 bf16 basis arrays (256x256 each) + softplus weights
  bf16* Wbr   = (bf16*)d_ws;
  bf16* Wbi   = Wbr + 65536;
  bf16* Wbin  = Wbi + 65536;
  bf16* WbrT  = Wbin + 65536;
  bf16* WbiT  = WbrT + 65536;
  bf16* WbinT = WbiT + 65536;
  float* Wf   = (float*)((char*)d_ws + 6 * 65536 * sizeof(bf16));

  qc_prep<<<256, 256, 0, stream>>>(basis_real, basis_imag, povm,
                                   Wbr, Wbi, Wbin, WbrT, WbiT, WbinT, Wf);

  const int blocks = (int)(rows / 64);  // 64 rows per block (4 waves x 16 rows)
  qc_main<<<blocks, 128, 0, stream>>>(x_real, x_imag,
                                      Wbr, Wbi, Wbin, WbrT, WbiT, WbinT, Wf,
                                      (float*)d_out, plane);
}
